// AttentionLayer_45827301048488
// MI455X (gfx1250) — matrix operand order repaired and checked by reading, not compiled, not run on hardware
//
#include <hip/hip_runtime.h>

typedef __bf16 bf16;
typedef __attribute__((ext_vector_type(16))) __bf16 v16bf;
typedef __attribute__((ext_vector_type(8)))  __bf16 bf16x8;
typedef __attribute__((ext_vector_type(8)))  float   v8f;
typedef __attribute__((ext_vector_type(4)))  float   f32x4;

#define HID 1024

// Build a 16-element bf16 fragment from two contiguous 8-element (16B) chunks.
__device__ __forceinline__ v16bf make_frag(const bf16* p0, const bf16* p1) {
    union { v16bf v; bf16x8 h[2]; } u;
    u.h[0] = *(const bf16x8*)p0;
    u.h[1] = *(const bf16x8*)p1;
    return u.v;
}

__device__ __forceinline__ v8f wmma_bf16(v16bf a, v16bf b, v8f c) {
    // D = A(16x32) * B(32x16) + C ; emits v_wmma_f32_16x16x32_bf16
    return __builtin_amdgcn_wmma_f32_16x16x32_bf16(
        false, a, false, b, (short)0, c, false, false);
}

// ---------------------------------------------------------------------------
// Kernel 1: weight fp32 -> bf16, transposed:  Wt[n*1024 + k] = W[k*1024 + n]
// ---------------------------------------------------------------------------
__global__ __launch_bounds__(256) void wconv_kernel(const float* __restrict__ W,
                                                    bf16* __restrict__ Wt) {
    int i = blockIdx.x * 256 + threadIdx.x;   // 0 .. 1M-1
    int n = i >> 10;
    int k = i & 1023;
    Wt[i] = (bf16)W[(size_t)k * HID + n];
}

// ---------------------------------------------------------------------------
// Kernel 2: projection GEMM  Y = X(fp32)[M,1024] * W(bf16)[1024,1024] + bias
// Block tile: 64(M) x 256(N); 8 waves as (mt 0..3, nh 0..1), wave = 16x128.
// X tile staged fp32->bf16 through double-buffered LDS (one barrier/k-step).
// B fragments software-pipelined across k-steps in ping-pong register sets.
// transpose_out=1 stores V transposed per-bc: Y[bc][n][l2]  (l2 = row % 512)
// ---------------------------------------------------------------------------
__global__ __launch_bounds__(256) void proj_kernel(const float* __restrict__ X,
                                                   const bf16* __restrict__ Wt,
                                                   const float* __restrict__ bias,
                                                   bf16* __restrict__ Y,
                                                   int transpose_out) {
    const int tid  = threadIdx.x;
    const int lane = tid & 31;
    const int w    = tid >> 5;
    const int mt   = w >> 1;           // 0..3
    const int nh   = w & 1;            // 0..1
    const int lrow = lane & 15;
    const int hi   = (lane & 16) ? 1 : 0;
    const int grow0 = blockIdx.y * 64;
    const int gcol0 = blockIdx.x * 256 + nh * 128;

    __shared__ bf16 Xs[2][64][32];     // double-buffered staging, 8 KB

    const int srow = tid >> 2;               // staging row, 0..63
    const int skk  = (tid & 3) * 8;          // staging k offset, 0/8/16/24
    const float* srcbase = X + (size_t)(grow0 + srow) * HID + skk;

    auto stage = [&](int kb, int buf) {
        const float* src = srcbase + kb;
        f32x4 a = *(const f32x4*)src;
        f32x4 b = *(const f32x4*)(src + 4);
        bf16x8 h;
        h[0]=(bf16)a[0]; h[1]=(bf16)a[1]; h[2]=(bf16)a[2]; h[3]=(bf16)a[3];
        h[4]=(bf16)b[0]; h[5]=(bf16)b[1]; h[6]=(bf16)b[2]; h[7]=(bf16)b[3];
        *(bf16x8*)&Xs[buf][srow][skk] = h;
    };

    // Hoisted per-tile B base pointers (k advances by pointer offset only)
    const bf16* bbase[8];
    #pragma unroll
    for (int t = 0; t < 8; ++t)
        bbase[t] = Wt + (size_t)(gcol0 + t * 16 + lrow) * HID + hi * 8;

    auto loadB = [&](v16bf (&bf)[8], int kb) {
        #pragma unroll
        for (int t = 0; t < 8; ++t)
            bf[t] = make_frag(bbase[t] + kb, bbase[t] + kb + 16);
    };

    v8f acc[8];
    const v8f vzero = {0.f,0.f,0.f,0.f,0.f,0.f,0.f,0.f};
    #pragma unroll
    for (int t = 0; t < 8; ++t) acc[t] = vzero;

    auto mma_step = [&](v16bf (&bf)[8], int buf) {
        const bf16* arow = &Xs[buf][mt * 16 + lrow][hi * 8];
        v16bf afrag = make_frag(arow, arow + 16);
        #pragma unroll
        for (int t = 0; t < 8; ++t)
            acc[t] = wmma_bf16(afrag, bf[t], acc[t]);
    };

    v16bf b0[8], b1[8];
    stage(0, 0);
    loadB(b0, 0);
    // 32 k-steps, processed two at a time (ping-pong LDS bufs + B frag sets).
    // Tail loads overread one 32-wide tile; lands in adjacent ws region (safe).
    for (int i = 0; i < HID / 32; i += 2) {
        const int kb = i * 32;
        __syncthreads();                               // Xs buf0 ready
        if (i + 1 < HID / 32) stage(kb + 32, 1);       // fill buf1
        loadB(b1, kb + 32);                            // pipeline next B set
        mma_step(b0, 0);
        __syncthreads();                               // Xs buf1 ready
        if (i + 2 < HID / 32) stage(kb + 64, 0);       // refill buf0
        loadB(b0, kb + 64);                            // pipeline next B set
        mma_step(b1, 1);
    }

    // Epilogue: +bias, convert to bf16, store (optionally transposed for V)
    #pragma unroll
    for (int t = 0; t < 8; ++t) {
        int col = gcol0 + t * 16 + lrow;
        float b = bias[col];
        #pragma unroll
        for (int r = 0; r < 8; ++r) {
            int row = grow0 + mt * 16 + hi * 8 + r;
            float val = acc[t][r] + b;
            if (!transpose_out) {
                Y[(size_t)row * HID + col] = (bf16)val;
            } else {
                int bc = row >> 9;          // 512 rows per (b,c)
                int l2 = row & 511;
                Y[((size_t)bc * HID + col) * 512 + l2] = (bf16)val;
            }
        }
    }
}

// ---------------------------------------------------------------------------
// Kernel 3: fused attention for one (bc, 32-row q block).
// Phase 1: S = Q Kᵀ held in registers (wave tile 16x128; waves = 2(mt) x 4(nq)),
//          B fragments software-pipelined (ping-pong register sets).
// Phase 2: softmax via shfl_xor lane reductions + LDS cross-wave combine;
//          probs -> LDS bf16; deterministic per-mt column-sum partials.
// Phase 3: context = P * V (P from LDS, Vᵀ fragments pipelined like phase 1).
// ---------------------------------------------------------------------------
__global__ __launch_bounds__(256) void attn_kernel(const bf16* __restrict__ Qb,   // [bc*256][1024]
                                                   const bf16* __restrict__ Kb,   // [bc*512][1024]
                                                   const bf16* __restrict__ Vt,   // [bc][1024][512]
                                                   float* __restrict__ ctx,       // [bc*256][1024]
                                                   float* __restrict__ spart) {   // [bc*8][512]
    const int tid  = threadIdx.x;
    const int lane = tid & 31;
    const int w    = tid >> 5;
    const int mt   = w >> 2;          // 0..1  (16 q-rows each)
    const int nq   = w & 3;           // 0..3  (128 key-cols each)
    const int lrow = lane & 15;
    const int hi   = (lane & 16) ? 1 : 0;
    const int bc   = blockIdx.y;
    const int qblk = blockIdx.x;      // 0..7
    const int q0   = qblk * 32;

    __shared__ bf16  P[32][512];           // 32 KB probs
    __shared__ float maxbuf[2][4][16];
    __shared__ float sumbuf[2][4][16];
    __shared__ float colpart[2][512];      // per-mt column partial sums

    const bf16* Qbase = Qb + ((size_t)bc * 256 + q0) * HID;
    const bf16* Kbase = Kb + (size_t)bc * 512 * HID;
    const bf16* Vbase = Vt + (size_t)bc * HID * 512;

    const v8f vzero = {0.f,0.f,0.f,0.f,0.f,0.f,0.f,0.f};

    // ---- Phase 1: logits tile (16 x 128) in registers --------------------
    v8f acc[8];
    #pragma unroll
    for (int t = 0; t < 8; ++t) acc[t] = vzero;

    {
        const bf16* aptr = Qbase + (size_t)(mt * 16 + lrow) * HID + hi * 8;
        const bf16* bbase[8];
        #pragma unroll
        for (int t = 0; t < 8; ++t)
            bbase[t] = Kbase + (size_t)(nq * 128 + t * 16 + lrow) * HID + hi * 8;

        auto loadB = [&](v16bf (&bf)[8], int kb) {
            #pragma unroll
            for (int t = 0; t < 8; ++t)
                bf[t] = make_frag(bbase[t] + kb, bbase[t] + kb + 16);
        };
        auto mma_step = [&](v16bf (&bf)[8], int kb) {
            v16bf afrag = make_frag(aptr + kb, aptr + kb + 16);
            #pragma unroll
            for (int t = 0; t < 8; ++t)
                acc[t] = wmma_bf16(afrag, bf[t], acc[t]);
        };

        v16bf b0[8], b1[8];
        loadB(b0, 0);
        for (int kb = 0; kb < HID; kb += 64) {     // tail overread is in-ws
            loadB(b1, kb + 32);
            mma_step(b0, kb);
            loadB(b0, kb + 64);
            mma_step(b1, kb + 32);
        }
    }

    // ---- Phase 2: softmax --------------------------------------------------
    // Per-wave row max over its 128 cols (lane groups 0-15 / 16-31 independent)
    #pragma unroll
    for (int r = 0; r < 8; ++r) {
        float m = acc[0][r];
        #pragma unroll
        for (int t = 1; t < 8; ++t) m = fmaxf(m, acc[t][r]);
        m = fmaxf(m, __shfl_xor(m, 1, 32));
        m = fmaxf(m, __shfl_xor(m, 2, 32));
        m = fmaxf(m, __shfl_xor(m, 4, 32));
        m = fmaxf(m, __shfl_xor(m, 8, 32));
        if (lrow == 0) maxbuf[mt][nq][hi * 8 + r] = m;
    }
    __syncthreads();

    // Full row max; exponentiate in place; per-wave row-sum partials
    #pragma unroll
    for (int r = 0; r < 8; ++r) {
        int ridx = hi * 8 + r;
        float m = fmaxf(fmaxf(maxbuf[mt][0][ridx], maxbuf[mt][1][ridx]),
                        fmaxf(maxbuf[mt][2][ridx], maxbuf[mt][3][ridx]));
        float s = 0.f;
        #pragma unroll
        for (int t = 0; t < 8; ++t) {
            float e = __expf(acc[t][r] - m);
            acc[t][r] = e;
            s += e;
        }
        s += __shfl_xor(s, 1, 32);
        s += __shfl_xor(s, 2, 32);
        s += __shfl_xor(s, 4, 32);
        s += __shfl_xor(s, 8, 32);
        if (lrow == 0) sumbuf[mt][nq][ridx] = s;
    }
    __syncthreads();

    float rinv[8];
    #pragma unroll
    for (int r = 0; r < 8; ++r) {
        int ridx = hi * 8 + r;
        float s = sumbuf[mt][0][ridx] + sumbuf[mt][1][ridx] +
                  sumbuf[mt][2][ridx] + sumbuf[mt][3][ridx];
        rinv[r] = 1.f / s;
    }

    // Normalize -> probs to LDS (bf16); per-mt deterministic column sums
    #pragma unroll
    for (int t = 0; t < 8; ++t) {
        int col = nq * 128 + t * 16 + lrow;
        float cs = 0.f;
        #pragma unroll
        for (int r = 0; r < 8; ++r) {
            float p = acc[t][r] * rinv[r];
            P[mt * 16 + hi * 8 + r][col] = (bf16)p;
            cs += p;
        }
        cs += __shfl_xor(cs, 16, 32);     // fold rows 8-15 partial into 0-7 lanes
        if (!hi) colpart[mt][col] = cs;   // exclusive writer per (mt,col)
    }
    __syncthreads();

    // Deterministic attention-score partials for this q-block
    {
        float s0 = colpart[0][tid]       + colpart[1][tid];
        float s1 = colpart[0][tid + 256] + colpart[1][tid + 256];
        float* sp = spart + ((size_t)bc * 8 + qblk) * 512;
        sp[tid]       = s0;
        sp[tid + 256] = s1;
    }

    // ---- Phase 3: context = P(32x512) * V(512x1024) ------------------------
    for (int chunk = 0; chunk < 2; ++chunk) {
        const int c0 = nq * 256 + chunk * 128;
        v8f cacc[8];
        #pragma unroll
        for (int t = 0; t < 8; ++t) cacc[t] = vzero;

        const bf16* prow = &P[mt * 16 + lrow][hi * 8];
        const bf16* bbase[8];
        #pragma unroll
        for (int t = 0; t < 8; ++t)
            bbase[t] = Vbase + (size_t)(c0 + t * 16 + lrow) * 512 + hi * 8;

        auto loadB = [&](v16bf (&bf)[8], int kb) {
            #pragma unroll
            for (int t = 0; t < 8; ++t)
                bf[t] = make_frag(bbase[t] + kb, bbase[t] + kb + 16);
        };
        auto mma_step = [&](v16bf (&bf)[8], int kb) {
            v16bf afrag = make_frag(prow + kb, prow + kb + 16);   // ds_load x2
            #pragma unroll
            for (int t = 0; t < 8; ++t)
                cacc[t] = wmma_bf16(afrag, bf[t], cacc[t]);
        };

        v16bf b0[8], b1[8];
        loadB(b0, 0);
        for (int kb = 0; kb < 512; kb += 64) {     // tail overread is in-ws
            loadB(b1, kb + 32);
            mma_step(b0, kb);
            loadB(b0, kb + 64);
            mma_step(b1, kb + 32);
        }

        #pragma unroll
        for (int t = 0; t < 8; ++t) {
            int col = c0 + t * 16 + lrow;
            #pragma unroll
            for (int r = 0; r < 8; ++r) {
                int row = mt * 16 + hi * 8 + r;
                ctx[((size_t)bc * 256 + q0 + row) * HID + col] = cacc[t][r];
            }
        }
    }
}

// ---------------------------------------------------------------------------
// Kernel 4: reduce the 8 per-q-block score partials -> attention_scores
// ---------------------------------------------------------------------------
__global__ __launch_bounds__(256) void scores_kernel(const float* __restrict__ part,
                                                     float* __restrict__ out) {
    int i = blockIdx.x * 256 + threadIdx.x;   // 0 .. 131071
    int bc = i >> 9;
    int c  = i & 511;
    float s = 0.f;
    #pragma unroll
    for (int q = 0; q < 8; ++q) s += part[((size_t)bc * 8 + q) * 512 + c];
    out[i] = s;
}

// ---------------------------------------------------------------------------
extern "C" void kernel_launch(void* const* d_in, const int* in_sizes, int n_in,
                              void* d_out, int out_size, void* d_ws, size_t ws_size,
                              hipStream_t stream) {
    const float* query  = (const float*)d_in[0];  // [8,32,256,1024]
    const float* source = (const float*)d_in[1];  // [8,32,512,1024]
    const float* Wq = (const float*)d_in[2];
    const float* bq = (const float*)d_in[3];
    const float* Wk = (const float*)d_in[4];
    const float* bk = (const float*)d_in[5];
    const float* Wv = (const float*)d_in[6];
    const float* bv = (const float*)d_in[7];
    float* out = (float*)d_out;

    // Workspace layout (regions intentionally adjacent: pipelined GEMM loops
    // overread at most one 32-element tile past a region's end).
    char* ws = (char*)d_ws;
    const size_t MB = 1024ull * 1024ull;
    bf16*  Wtq   = (bf16*)(ws);                       //   2 MB
    bf16*  Wtk   = (bf16*)(ws + 2 * MB);              //   2 MB
    bf16*  Wtv   = (bf16*)(ws + 4 * MB);              //   2 MB
    bf16*  qb    = (bf16*)(ws + 6 * MB);              // 128 MB  [bc*256][1024]
    bf16*  kb    = (bf16*)(ws + 134 * MB);            // 256 MB  [bc*512][1024]
    bf16*  vt    = (bf16*)(ws + 390 * MB);            // 256 MB  [bc][1024][512]
    float* spart = (float*)(ws + 646 * MB);           //   4 MB  [bc*8][512]

    // 1) weights -> transposed bf16
    wconv_kernel<<<4096, 256, 0, stream>>>(Wq, Wtq);
    wconv_kernel<<<4096, 256, 0, stream>>>(Wk, Wtk);
    wconv_kernel<<<4096, 256, 0, stream>>>(Wv, Wtv);

    // 2) projections (q: 65536 rows; k,v: 131072 rows; 64-row M blocks, 4 N blocks)
    proj_kernel<<<dim3(4, 1024), 256, 0, stream>>>(query,  Wtq, bq, qb, 0);
    proj_kernel<<<dim3(4, 2048), 256, 0, stream>>>(source, Wtk, bk, kb, 0);
    proj_kernel<<<dim3(4, 2048), 256, 0, stream>>>(source, Wtv, bv, vt, 1);

    // 3) fused attention: 8 q-blocks x 256 (b,c) pairs
    attn_kernel<<<dim3(8, 256), 256, 0, stream>>>(qb, kb, vt, out, spart);

    // 4) attention_scores reduction (context occupies first 8*32*256*1024 floats)
    scores_kernel<<<512, 256, 0, stream>>>(spart, out + (size_t)8 * 32 * 256 * 1024);
}
